// Sequence_34127810134257
// MI455X (gfx1250) — matrix operand order repaired and checked by reading, not compiled, not run on hardware
//
#include <hip/hip_runtime.h>
#include <hip/hip_bf16.h>

// ---------------------------------------------------------------------------
// 2-layer LSTM (T=4096, M=512, D=11, N=51) + linear head, CDNA5 wave32 WMMA.
// Batch is partitioned into 32 independent 16-row tiles (one workgroup each);
// the 4096-step recurrence runs inside the kernel with 4 barriers/step.
// GEMMs use v_wmma_f32_16x16x32_f16 (f16 in, fp32 accumulate); cell state and
// activations stay fp32 (v_exp_f32 + v_rcp_f32, no IEEE div). Weight (B)
// fragments and per-tile biases are register-resident for the whole loop
// (>256 VGPRs via VGPR-MSB). Layer-2's h2-dependent WMMAs are hoisted into
// phase A (they don't depend on pointwise-1), with partial accumulators kept
// in registers across the barrier.
// ---------------------------------------------------------------------------

#define T_STEPS 4096
#define M_BATCH 512
#define DIN     11
#define N_HID   51
#define G4      204   // 4*N gate width
#define GP      224   // padded gate width (13 tiles of 16 + zero pad to 224)
#define NT      13    // number of 16-wide N tiles
#define KP      64    // padded hidden K (51 -> 64)
#define KX      32    // padded input K (11 -> 32)

typedef __attribute__((ext_vector_type(16))) _Float16 v16h;
typedef __attribute__((ext_vector_type(8)))  _Float16 v8h;
typedef __attribute__((ext_vector_type(8)))  float    v8f;

__device__ __forceinline__ v8f wmma_f16(v16h a, v16h b, v8f c) {
  // (neg_a, A, neg_b, B, c_mod, C, reuse_a, reuse_b)
  return __builtin_amdgcn_wmma_f32_16x16x32_f16(false, a, false, b, (short)0, c,
                                                false, false);
}

__device__ __forceinline__ v8f splat8(float x) {
  v8f r = {x, x, x, x, x, x, x, x};
  return r;
}

// A-matrix (16x32 f16) fragment from LDS row-major [16][kpad].
// ISA layout: lanes 0-15 row M=lane hold K = kbase+0..7 (VGPR0-3) and
// kbase+16..23 (VGPR4-7); lanes 16-31 hold K = kbase+8..15 / kbase+24..31.
__device__ __forceinline__ v16h load_afrag(const _Float16* base, int kpad,
                                           int kbase, int lane) {
  const _Float16* r = base + (lane & 15) * kpad + kbase + ((lane >> 4) << 3);
  v8h lo = *(const v8h*)(r);
  v8h hi = *(const v8h*)(r + 16);
  return __builtin_shufflevector(lo, hi, 0, 1, 2, 3, 4, 5, 6, 7,
                                 8, 9, 10, 11, 12, 13, 14, 15);
}

// B-matrix (32x16 f16) fragment built directly from global fp32 weights W
// (shape [204][kdim], row-major; gates = h @ W^T so B[k][n] = W[n][k]).
// ISA layout: lane col = lane&15, element j holds K = 8*(lane>>4) + j (j<8),
// 16 + 8*(lane>>4) + (j-8) (j>=8).
__device__ __forceinline__ v16h make_bfrag(const float* __restrict__ W,
                                           int kdim, int krows, int nt, int kh,
                                           int lane) {
  const int col   = nt * 16 + (lane & 15);
  const int kbase = 32 * kh + 8 * (lane >> 4);
  v16h b;
#pragma unroll
  for (int j = 0; j < 16; ++j) {
    const int k = kbase + j + ((j >> 3) << 3);
    const float v = (col < G4 && k < krows) ? W[col * kdim + k] : 0.0f;
    b[j] = (_Float16)v;
  }
  return b;
}

// Fast activations: v_exp_f32 + v_rcp_f32, overflow-safe at +-inf.
__device__ __forceinline__ float fast_rcp(float x) {
  return __builtin_amdgcn_rcpf(x);
}
__device__ __forceinline__ float sigm_f(float x) {
  return fast_rcp(1.0f + __expf(-x));      // exp(-x)->inf => rcp(inf)=0
}
__device__ __forceinline__ float tanh_f(float x) {
  return 1.0f - 2.0f * fast_rcp(__expf(2.0f * x) + 1.0f);  // -> +-1 at +-inf
}

__global__ __launch_bounds__(128, 1)
void lstm2_wmma_kernel(const float* __restrict__ input,
                       const float* __restrict__ W_ih1,
                       const float* __restrict__ W_hh1,
                       const float* __restrict__ b_ih1,
                       const float* __restrict__ b_hh1,
                       const float* __restrict__ W_ih2,
                       const float* __restrict__ W_hh2,
                       const float* __restrict__ b_ih2,
                       const float* __restrict__ b_hh2,
                       const float* __restrict__ W_lin,
                       const float* __restrict__ b_lin,
                       float* __restrict__ out) {
  __shared__ __align__(16) _Float16 xs[16][KX];    // x_t tile (f16, K-padded)
  __shared__ __align__(16) _Float16 h1s[16][KP];   // h1 (f16, K-padded to 0)
  __shared__ __align__(16) _Float16 h2s[16][KP];   // h2 (f16, K-padded to 0)
  __shared__ __align__(16) float    gates[16][GP]; // gate pre-acts (pad = 0)
  __shared__ __align__(16) float    c1s[16][KP];   // fp32 cell (pad = 0)
  __shared__ __align__(16) float    c2s[16][KP];
  __shared__ __align__(16) float    h2f[16][KP];   // fp32 h2 (pad = 0)
  __shared__ float gb1[G4 + 4], gb2[G4 + 4];       // fused biases (init only)
  __shared__ __align__(16) float wl[52];           // W_lin padded, wl[51]=0
  __shared__ float blin;

  const int tid  = threadIdx.x;
  const int w    = tid >> 5;    // wave id (wave32)
  const int lane = tid & 31;
  const int half = lane >> 4;
  const int colr = lane & 15;
  const int m0   = blockIdx.x * 16;
  // Rotated tile bases: wave 0 (head duty) gets 3 tiles, wave 3 gets 4.
  const int ntbase = (w + 1) & 3;

  // ---- one-time init --------------------------------------------------------
  for (int i = tid; i < 16 * KX; i += 128) (&xs[0][0])[i] = (_Float16)0.0f;
  for (int i = tid; i < 16 * KP; i += 128) {
    (&h1s[0][0])[i] = (_Float16)0.0f;
    (&h2s[0][0])[i] = (_Float16)0.0f;
    (&c1s[0][0])[i] = 0.0f;
    (&c2s[0][0])[i] = 0.0f;
    (&h2f[0][0])[i] = 0.0f;
  }
  for (int i = tid; i < 16 * GP; i += 128) (&gates[0][0])[i] = 0.0f;
  for (int g = tid; g < G4 + 4; g += 128) {
    gb1[g] = (g < G4) ? (b_ih1[g] + b_hh1[g]) : 0.0f;
    gb2[g] = (g < G4) ? (b_ih2[g] + b_hh2[g]) : 0.0f;
  }
  for (int j = tid; j < 52; j += 128) wl[j] = (j < N_HID) ? W_lin[j] : 0.0f;
  if (tid == 0) blin = b_lin[0];

  // Register-resident B fragments: per wave up to 4 N-tiles x 7 fragments.
  v16h bIH1[4], bHH1[2][4], bIH2[2][4], bHH2[2][4];
#pragma unroll
  for (int i = 0; i < 4; ++i) {
    const int nt = ntbase + 4 * i;                 // covers tiles 0..12
    bIH1[i]    = make_bfrag(W_ih1, DIN,   DIN,   nt, 0, lane);
    bHH1[0][i] = make_bfrag(W_hh1, N_HID, N_HID, nt, 0, lane);
    bHH1[1][i] = make_bfrag(W_hh1, N_HID, N_HID, nt, 1, lane);
    bIH2[0][i] = make_bfrag(W_ih2, N_HID, N_HID, nt, 0, lane);
    bIH2[1][i] = make_bfrag(W_ih2, N_HID, N_HID, nt, 1, lane);
    bHH2[0][i] = make_bfrag(W_hh2, N_HID, N_HID, nt, 0, lane);
    bHH2[1][i] = make_bfrag(W_hh2, N_HID, N_HID, nt, 1, lane);
  }

  // stage x_0 (division-free enumeration over padded 16x16 space)
  {
    const float* xin = input + (size_t)m0 * DIN;
#pragma unroll
    for (int it = 0; it < 2; ++it) {
      const int e = tid + it * 128;              // 0..255
      const int m = e >> 4, k = e & 15;
      if (k < DIN) xs[m][k] = (_Float16)xin[m * DIN + k];
    }
  }
  __syncthreads();

  // Hoist time-invariant per-tile biases into registers (no in-loop ds_load).
  float bias1r[4], bias2r[4];
#pragma unroll
  for (int i = 0; i < 4; ++i) {
    const int nt = ntbase + 4 * i;
    const int c = (nt < NT) ? (nt * 16 + colr) : 0;
    bias1r[i] = gb1[c];
    bias2r[i] = gb2[c];
  }

  // ---- sequential time loop (4 barriers per step) ---------------------------
  for (int t = 0; t < T_STEPS; ++t) {
    v8f acc2[4];   // layer-2 partials, live across the phase-A/C barriers

    // ---- phase A: layer-1 gates (x,h1) + layer-2 h2-partials ----
    {
      const v16h ax  = load_afrag(&xs[0][0], KX, 0, lane);
      const v16h a0  = load_afrag(&h1s[0][0], KP, 0, lane);
      const v16h a1  = load_afrag(&h1s[0][0], KP, 32, lane);
      const v16h b0  = load_afrag(&h2s[0][0], KP, 0, lane);
      const v16h b1  = load_afrag(&h2s[0][0], KP, 32, lane);
#pragma unroll
      for (int i = 0; i < 4; ++i) {
        const int nt = ntbase + 4 * i;
        if (nt < NT) {
          v8f acc = splat8(bias1r[i]);
          acc = wmma_f16(ax, bIH1[i], acc);
          acc = wmma_f16(a0, bHH1[0][i], acc);
          acc = wmma_f16(a1, bHH1[1][i], acc);
#pragma unroll
          for (int v = 0; v < 8; ++v)
            gates[v + 8 * half][nt * 16 + colr] = acc[v];
          v8f p = splat8(bias2r[i]);
          p = wmma_f16(b0, bHH2[0][i], p);
          p = wmma_f16(b1, bHH2[1][i], p);
          acc2[i] = p;
        }
      }
    }
    __syncthreads();

    // ---- phase B: layer-1 pointwise (branch-free; pad lanes masked to 0) ----
#pragma unroll
    for (int it = 0; it < 8; ++it) {
      const int e = tid + it * 128;              // 0..1023 over [16][64]
      const int m = e >> 6, j = e & 63;
      const float keep = (j < N_HID) ? 1.0f : 0.0f;
      const float gi = gates[m][j];
      const float gf = gates[m][j + N_HID];
      const float gg = gates[m][j + 2 * N_HID];
      const float go = gates[m][j + 3 * N_HID];
      const float cn = keep * (sigm_f(gf) * c1s[m][j] + sigm_f(gi) * tanh_f(gg));
      const float hn = keep * (sigm_f(go) * tanh_f(cn));
      c1s[m][j] = cn;
      h1s[m][j] = (_Float16)hn;
    }
    __syncthreads();

    // ---- phase C: layer-2 gates: add h1-part onto register partials ----
    {
      const v16h a10 = load_afrag(&h1s[0][0], KP, 0, lane);
      const v16h a11 = load_afrag(&h1s[0][0], KP, 32, lane);
#pragma unroll
      for (int i = 0; i < 4; ++i) {
        const int nt = ntbase + 4 * i;
        if (nt < NT) {
          v8f acc = acc2[i];
          acc = wmma_f16(a10, bIH2[0][i], acc);
          acc = wmma_f16(a11, bIH2[1][i], acc);
#pragma unroll
          for (int v = 0; v < 8; ++v)
            gates[v + 8 * half][nt * 16 + colr] = acc[v];
        }
      }
    }
    __syncthreads();

    // ---- phase D: layer-2 pointwise + stage x_{t+1} ----
#pragma unroll
    for (int it = 0; it < 8; ++it) {
      const int e = tid + it * 128;
      const int m = e >> 6, j = e & 63;
      const float keep = (j < N_HID) ? 1.0f : 0.0f;
      const float gi = gates[m][j];
      const float gf = gates[m][j + N_HID];
      const float gg = gates[m][j + 2 * N_HID];
      const float go = gates[m][j + 3 * N_HID];
      const float cn = keep * (sigm_f(gf) * c2s[m][j] + sigm_f(gi) * tanh_f(gg));
      const float hn = keep * (sigm_f(go) * tanh_f(cn));
      c2s[m][j] = cn;
      h2s[m][j] = (_Float16)hn;
      h2f[m][j] = hn;
    }
    if (t + 1 < T_STEPS) {
      const float* xin =
          input + (size_t)(t + 1) * (M_BATCH * DIN) + (size_t)m0 * DIN;
#pragma unroll
      for (int it = 0; it < 2; ++it) {
        const int e = tid + it * 128;
        const int m = e >> 4, k = e & 15;
        if (k < DIN) xs[m][k] = (_Float16)xin[m * DIN + k];
      }
      if (lane == 0 && w < 3 && (t + 2) < T_STEPS)
        __builtin_prefetch(xin + M_BATCH * DIN + w * 64, 0, 0);
    }
    __syncthreads();

    // ---- linear head: out[m, t] (wave 0 only; overlaps next step's phase A
    //      in waves 1-3, no trailing barrier). 13x float4 dual loads. ----
    if (tid < 16) {
      const float4* hv = (const float4*)(&h2f[tid][0]);   // 256B row stride
      const float4* wv = (const float4*)(&wl[0]);
      float acc = blin;
#pragma unroll
      for (int q = 0; q < 13; ++q) {
        const float4 a = hv[q];
        const float4 b = wv[q];
        acc += a.x * b.x + a.y * b.y + a.z * b.z + a.w * b.w;
      }
      out[(size_t)(m0 + tid) * T_STEPS + t] = acc;
    }
  }
}

extern "C" void kernel_launch(void* const* d_in, const int* in_sizes, int n_in,
                              void* d_out, int out_size, void* d_ws,
                              size_t ws_size, hipStream_t stream) {
  (void)in_sizes; (void)n_in; (void)out_size; (void)d_ws; (void)ws_size;
  const float* input = (const float*)d_in[0];
  const float* W_ih1 = (const float*)d_in[1];
  const float* W_hh1 = (const float*)d_in[2];
  const float* b_ih1 = (const float*)d_in[3];
  const float* b_hh1 = (const float*)d_in[4];
  const float* W_ih2 = (const float*)d_in[5];
  const float* W_hh2 = (const float*)d_in[6];
  const float* b_ih2 = (const float*)d_in[7];
  const float* b_hh2 = (const float*)d_in[8];
  const float* W_lin = (const float*)d_in[9];
  const float* b_lin = (const float*)d_in[10];
  float* out = (float*)d_out;

  lstm2_wmma_kernel<<<dim3(M_BATCH / 16), dim3(128), 0, stream>>>(
      input, W_ih1, W_hh1, b_ih1, b_hh1, W_ih2, W_hh2, b_ih2, b_hh2, W_lin,
      b_lin, out);
}
